// nh_Block_51213190037564
// MI455X (gfx1250) — matrix operand order repaired and checked by reading, not compiled, not run on hardware
//
#include <hip/hip_runtime.h>
#include <math.h>

// ---------------------------------------------------------------------------
// Problem constants (match the reference)
// ---------------------------------------------------------------------------
namespace {
constexpr int B_   = 2;
constexpr int S_   = 8192;
constexpr int T_   = 4096;
constexpr int KNN  = 16;            // NH neighbors
constexpr int MD   = 64;            // model dim
constexpr int FF   = 128;           // ff dim
constexpr int H_   = 4;             // heads
constexpr int HD   = 16;            // head dim
constexpr int NCH  = 16;            // s-chunks for 2-stage KNN
constexpr int CHUNK = S_ / NCH;     // 512
constexpr int TOK  = B_ * T_;       // 8192 tokens
constexpr int R_   = TOK * KNN;     // 131072 rows (token,neighbor)
constexpr float LOG100 = 4.605170185988092f;
}

// ---------------------------------------------------------------------------
// WMMA bf16 plumbing (CDNA5 / gfx1250, wave32)
// ---------------------------------------------------------------------------
typedef __attribute__((ext_vector_type(16))) __bf16 v16bf;
typedef __attribute__((ext_vector_type(8)))  __bf16 v8bf;
typedef __attribute__((ext_vector_type(8)))  float  v8f;
typedef __attribute__((ext_vector_type(4)))  unsigned int u32x4;
typedef __attribute__((ext_vector_type(8)))  int i32x8;
typedef __attribute__((ext_vector_type(4)))  int i32x4;

__device__ __forceinline__ v8f wmma_bf16(v16bf a, v16bf b, v8f c) {
  // D = A(16x32) * B(32x16) + C(16x16), f32 accumulate
  return __builtin_amdgcn_wmma_f32_16x16x32_bf16(false, a, false, b,
                                                 (short)0, c, false, false);
}

// A fragment 16x32 from row-major bf16 (global or LDS), row stride ldk.
// ISA layout: lane m=l&15, hi=l>>4; element e -> K = (e>>3)*16 + hi*8 + (e&7)
// => two contiguous 8-half (16 B) runs per lane: [kbase+hi*8, +8) and +16.
__device__ __forceinline__ v16bf load_a_bf16(const __bf16* base, int kbase, int ldk) {
  const int l  = threadIdx.x & 31;
  const int m  = l & 15;
  const int hi = l >> 4;
  const __bf16* p = base + (size_t)m * ldk + kbase + hi * 8;
  const v8bf lo = *(const v8bf*)p;        // K = kbase + hi*8 + [0..7]
  const v8bf hh = *(const v8bf*)(p + 16); // K = kbase + 16 + hi*8 + [0..7]
  return __builtin_shufflevector(lo, hh, 0, 1, 2, 3, 4, 5, 6, 7,
                                 8, 9, 10, 11, 12, 13, 14, 15);
}

// B fragments come from weights pre-swizzled into fragment order:
// per 32x16 (KxN) tile, lane l owns 16 contiguous halfs (32 B aligned).
__device__ __forceinline__ v16bf load_b_packed(const __bf16* Wp, int ktile, int ntile,
                                               int ntiles) {
  const int l = threadIdx.x & 31;
  return *(const v16bf*)(Wp + ((size_t)ktile * ntiles + ntile) * 512 + l * 16);
}

// C/D: v8f element r -> row M = r + 8*hi, col N = l&15
template <int ACT /*0=none,1=lrelu*/, int OUTBF /*0=f32,1=bf16*/>
__device__ __forceinline__ void store_tile(void* out, int ldn, int row0, int nbase,
                                           v8f c, const float* bias) {
  const int l  = threadIdx.x & 31;
  const int n  = l & 15;
  const int hi = l >> 4;
  const float bv = bias ? bias[nbase + n] : 0.0f;
#pragma unroll
  for (int r = 0; r < 8; ++r) {
    const int m = r + 8 * hi;
    float v = c[r] + bv;
    if (ACT == 1) v = (v >= 0.0f) ? v : 0.2f * v;
    const size_t o = (size_t)(row0 + m) * ldn + nbase + n;
    if (OUTBF) ((__bf16*)out)[o] = (__bf16)v;
    else       ((float*)out)[o]  = v;
  }
}

// ---------------------------------------------------------------------------
// Weight conversion f32 -> bf16 *fragment-packed* (one-time, <=2 MB each):
// dst[(ktile*(N/16)+ntile)*512 + lane*16 + e] with lane = hi*16 + (n&15),
// hi = (k>>3)&1, e = (k&7) + ((k>>4)&1)*8  (element e -> K = (e>>3)*16 + hi*8 + (e&7))
// ---------------------------------------------------------------------------
__global__ void pack_b_bf16(const float* __restrict__ in, __bf16* __restrict__ out,
                            int K, int N) {
  const int gid = blockIdx.x * blockDim.x + threadIdx.x;
  if (gid >= K * N) return;
  const int k = gid / N, n = gid % N;
  const int ktile = k >> 5, ntile = n >> 4;
  const int hi = (k >> 3) & 1, e = (k & 7) + ((k >> 4) & 1) * 8;
  const int lane = hi * 16 + (n & 15);
  out[((size_t)ktile * (N >> 4) + ntile) * 512 + lane * 16 + e] = (__bf16)in[gid];
}

// ---------------------------------------------------------------------------
// KNN stage 1: each thread scans one (b, t, chunk) slice of s, keeps top-16
// smallest d = c1^2 + c2^2, carrying (d, c1, c2, s). Coalesced over t.
// ---------------------------------------------------------------------------
__device__ __forceinline__ void topk_insert(float d, float a, float b, int s,
                                            float* dk, float* v1, float* v2, int* ik) {
#pragma unroll
  for (int j = 0; j < KNN; ++j) {
    if (d < dk[j]) {
      float t0 = dk[j]; dk[j] = d; d = t0;
      float t1 = v1[j]; v1[j] = a; a = t1;
      float t2 = v2[j]; v2[j] = b; b = t2;
      int   t3 = ik[j]; ik[j] = s; s = t3;
    }
  }
}

__global__ void knn_stage1(const float* __restrict__ c1, const float* __restrict__ c2,
                           float4* __restrict__ cand) {
  const int gid = blockIdx.x * blockDim.x + threadIdx.x;  // [0, B*T*NCH)
  const int t   = gid % T_;
  const int ch  = (gid / T_) % NCH;
  const int b   = gid / (T_ * NCH);
  float dk[KNN], v1k[KNN], v2k[KNN]; int ik[KNN];
#pragma unroll
  for (int j = 0; j < KNN; ++j) { dk[j] = 3.0e38f; v1k[j] = 0.f; v2k[j] = 0.f; ik[j] = 0; }
  const size_t plane = (size_t)b * S_ * T_;
  const int s0 = ch * CHUNK;
  for (int s = s0; s < s0 + CHUNK; ++s) {
    const size_t o = plane + (size_t)s * T_ + t;
    const float a = c1[o], bb = c2[o];
    const float d = a * a + bb * bb;
    if (d < dk[KNN - 1]) topk_insert(d, a, bb, s, dk, v1k, v2k, ik);
  }
  const size_t ob = (size_t)gid * KNN;  // layout [(b,ch,t), KNN]
#pragma unroll
  for (int j = 0; j < KNN; ++j)
    cand[ob + j] = make_float4(dk[j], v1k[j], v2k[j], __int_as_float(ik[j]));
}

// KNN stage 2: merge 16 sorted chunk-lists per token; gather x; emit
// nbr[tok*16+n] = {c1g, c2g, x_gathered, 0}
__global__ void knn_stage2(const float4* __restrict__ cand, const float* __restrict__ x,
                           float4* __restrict__ nbr) {
  const int tok = blockIdx.x * blockDim.x + threadIdx.x;  // [0, B*T)
  const int b = tok / T_, t = tok % T_;
  float dk[KNN], v1k[KNN], v2k[KNN]; int ik[KNN];
#pragma unroll
  for (int j = 0; j < KNN; ++j) { dk[j] = 3.0e38f; v1k[j] = 0.f; v2k[j] = 0.f; ik[j] = 0; }
  for (int ch = 0; ch < NCH; ++ch) {
    const size_t base = ((size_t)(b * NCH + ch) * T_ + t) * KNN;
    for (int j = 0; j < KNN; ++j) {            // chunk list is sorted ascending
      const float4 e = cand[base + j];
      if (!(e.x < dk[KNN - 1])) break;
      topk_insert(e.x, e.y, e.z, __float_as_int(e.w), dk, v1k, v2k, ik);
    }
  }
  const size_t ob = (size_t)tok * KNN;
#pragma unroll
  for (int n = 0; n < KNN; ++n)
    nbr[ob + n] = make_float4(v1k[n], v2k[n], x[(size_t)b * S_ + ik[n]], 0.f);
}

// ---------------------------------------------------------------------------
// PE layer 1: a1[r,f] = gelu(c1*W1[0,f] + c2*W1[1,f] + b1[f]) -> bf16
// ---------------------------------------------------------------------------
__global__ void pe1_kernel(const float4* __restrict__ nbr, const float* __restrict__ W1,
                           const float* __restrict__ b1, __bf16* __restrict__ a1) {
  const int gid = blockIdx.x * blockDim.x + threadIdx.x;  // [0, R*FF)
  const int f = gid & (FF - 1);
  const int r = gid >> 7;
  const float4 nb = nbr[r];
  const float z  = nb.x * W1[f] + nb.y * W1[FF + f] + b1[f];
  const float u  = 0.7978845608028654f * (z + 0.044715f * z * z * z);
  const float gg = 0.5f * z * (1.0f + tanhf(u));
  a1[(size_t)r * FF + f] = (__bf16)gg;
}

// ---------------------------------------------------------------------------
// Generic batched row-GEMM via WMMA: out[rows,N] = act(A[rows,K] @ W[K,N] + b)
// One wave per 16-row tile; NT = N/16 accumulator tiles per wave.
// A: row-major bf16 (2x b128 per fragment). W: fragment-packed bf16.
// ---------------------------------------------------------------------------
template <int K, int N, int ACT, int OUTBF>
__global__ __launch_bounds__(256) void rowgemm(const __bf16* __restrict__ A,
                                               const __bf16* __restrict__ W,
                                               const float* __restrict__ bias,
                                               void* __restrict__ out) {
  const int wave = threadIdx.x >> 5;
  const int row0 = (blockIdx.x * 8 + wave) * 16;
  constexpr int NT = N / 16;
  v8f acc[NT] = {};
  const __bf16* Arow = A + (size_t)row0 * K;
#pragma unroll
  for (int kt = 0; kt < K / 32; ++kt) {
    const v16bf a = load_a_bf16(Arow, kt * 32, K);
#pragma unroll
    for (int nt = 0; nt < NT; ++nt)
      acc[nt] = wmma_bf16(a, load_b_packed(W, kt, nt, NT), acc[nt]);
  }
#pragma unroll
  for (int nt = 0; nt < NT; ++nt)
    store_tile<ACT, OUTBF>(out, N, row0, nt * 16, acc[nt], bias);
}

// ---------------------------------------------------------------------------
// Big unfold-MLP GEMM: out[8192,1024] = act(A[8192,1024] @ W[1024,1024] + b)
// 16x1024 bf16 A tile staged into LDS by the Tensor Data Mover (one D#,
// 2-D tile, issued by wave 0, completed with s_wait_tensorcnt); 8 waves x
// 128-col spans of bf16 WMMA. W (2 MB packed bf16) stays hot in 192 MB L2.
// ---------------------------------------------------------------------------
template <int ACT, int OUTBF>
__global__ __launch_bounds__(256) void mu_gemm(const __bf16* __restrict__ A,
                                               const __bf16* __restrict__ W,
                                               const float* __restrict__ bias,
                                               void* __restrict__ out) {
  __shared__ __bf16 As[16 * 1024];   // 32 KB tile
  const int tid  = threadIdx.x;
  const int row0 = blockIdx.x * 16;
#if defined(__AMDGCN__)
  if (tid < 32) {  // one TDM descriptor per workgroup (EXEC ignored by TDM)
    const unsigned long long ga =
        (unsigned long long)(uintptr_t)(A + (size_t)row0 * 1024);
    const unsigned int lds = (unsigned int)(uintptr_t)As;
    u32x4 g0;
    g0[0] = 1u;                                   // count=1, user mode, no gather
    g0[1] = lds;                                  // lds_addr (bytes)
    g0[2] = (unsigned int)(ga & 0xFFFFFFFFu);     // global_addr[31:0]
    g0[3] = (unsigned int)((ga >> 32) & 0x01FFFFFFu) | (2u << 30);  // [56:32]|type=2
    i32x8 g1;
    g1[0] = (int)(1u << 16);                      // data_size=1 (2-byte elements)
    g1[1] = (int)((1024u & 0xFFFFu) << 16);       // tensor_dim0[15:0] in bits 63:48
    g1[2] = (int)((1024u >> 16) | (16u << 16));   // tensor_dim0[31:16] | tensor_dim1 lo
    g1[3] = (int)(1024u << 16);                   // tensor_dim1 hi | tile_dim0=1024
    g1[4] = 16;                                   // tile_dim1=16, tile_dim2=0
    g1[5] = 1024;                                 // tensor_dim0_stride lo32
    g1[6] = 0;
    g1[7] = 0;
    const i32x4 z4 = {0, 0, 0, 0};
#if __clang_major__ >= 23
    const i32x8 z8 = {0, 0, 0, 0, 0, 0, 0, 0};
    __builtin_amdgcn_tensor_load_to_lds(g0, g1, z4, z4, z8, 0);
#else
    __builtin_amdgcn_tensor_load_to_lds(g0, g1, z4, z4, 0);
#endif
    __builtin_amdgcn_s_wait_tensorcnt(0);
  }
#endif
  __syncthreads();
  const int wave = tid >> 5;
  v8f acc[8] = {};
  for (int kt = 0; kt < 32; ++kt) {
    if (kt + 1 < 32)
      __builtin_prefetch((const void*)(W + ((size_t)(kt + 1) * 64 + wave * 8) * 512), 0, 1);
    const v16bf a = load_a_bf16(As, kt * 32, 1024);
#pragma unroll
    for (int nt = 0; nt < 8; ++nt)
      acc[nt] = wmma_bf16(a, load_b_packed(W, kt, wave * 8 + nt, 64), acc[nt]);
  }
#pragma unroll
  for (int nt = 0; nt < 8; ++nt)
    store_tile<ACT, OUTBF>(out, 1024, row0, (wave * 8 + nt) * 16, acc[nt], bias);
}

// ---------------------------------------------------------------------------
// LayerNorms over MD=64 per row
// ---------------------------------------------------------------------------
__global__ void ln1_kernel(const float* __restrict__ pe, const float4* __restrict__ nbr,
                           const float* __restrict__ g, const float* __restrict__ be,
                           float* __restrict__ h, __bf16* __restrict__ hb) {
  const int r = blockIdx.x * blockDim.x + threadIdx.x;
  const float xv = nbr[r].z;
  float v[MD];
  const float4* p = (const float4*)(pe + (size_t)r * MD);
  float s = 0.f;
#pragma unroll
  for (int i = 0; i < MD / 4; ++i) {
    const float4 q4 = p[i];
    v[4 * i + 0] = q4.x + xv; v[4 * i + 1] = q4.y + xv;
    v[4 * i + 2] = q4.z + xv; v[4 * i + 3] = q4.w + xv;
    s += v[4 * i] + v[4 * i + 1] + v[4 * i + 2] + v[4 * i + 3];
  }
  const float m = s * (1.0f / MD);
  float ss = 0.f;
#pragma unroll
  for (int i = 0; i < MD; ++i) { const float d = v[i] - m; ss += d * d; }
  const float rstd = rsqrtf(ss * (1.0f / MD) + 1e-5f);
  const size_t ob = (size_t)r * MD;
#pragma unroll
  for (int i = 0; i < MD; ++i) {
    const float o = (v[i] - m) * rstd * g[i] + be[i];
    h[ob + i] = o; hb[ob + i] = (__bf16)o;
  }
}

__global__ void ln2_kernel(const float* __restrict__ hin, const float* __restrict__ ho,
                           const float* __restrict__ g, const float* __restrict__ be,
                           float* __restrict__ h2, __bf16* __restrict__ h2b) {
  const int r = blockIdx.x * blockDim.x + threadIdx.x;
  float v[MD];
  const size_t rb = (size_t)r * MD;
  float s = 0.f;
#pragma unroll
  for (int i = 0; i < MD; ++i) { v[i] = hin[rb + i] + ho[rb + i]; s += v[i]; }
  const float m = s * (1.0f / MD);
  float ss = 0.f;
#pragma unroll
  for (int i = 0; i < MD; ++i) { const float d = v[i] - m; ss += d * d; }
  const float rstd = rsqrtf(ss * (1.0f / MD) + 1e-5f);
#pragma unroll
  for (int i = 0; i < MD; ++i) {
    const float o = (v[i] - m) * rstd * g[i] + be[i];
    h2[rb + i] = o; h2b[rb + i] = (__bf16)o;
  }
}

// ---------------------------------------------------------------------------
// Cosine attention per token: one wave per token (wave32). V is rank-1
// (v[m,:] = x_m * Wv), so o[h,n,:] = (att[h,n,:] . x) * Wv_head.
// Lane l: neighbor n = l&15, head pair hp = l>>4 -> heads {2hp, 2hp+1}.
// ---------------------------------------------------------------------------
__global__ __launch_bounds__(256) void attn_kernel(const float* __restrict__ q,
                                                   const float* __restrict__ k,
                                                   const float4* __restrict__ nbr,
                                                   const float* __restrict__ Wv,
                                                   const float* __restrict__ logit_scale,
                                                   __bf16* __restrict__ o) {
  __shared__ float kn_lds[8][H_][KNN][HD];
  __shared__ float x_lds[8][KNN];
  const int wave = threadIdx.x >> 5;
  const int lane = threadIdx.x & 31;
  const int tok  = blockIdx.x * 8 + wave;
  const int n    = lane & 15;
  const int hp   = lane >> 4;
  const size_t rb = (size_t)tok * KNN;

  if (hp == 0) x_lds[wave][n] = nbr[rb + n].z;

  float qn[2][HD];
#pragma unroll
  for (int hh = 0; hh < 2; ++hh) {
    const int h = hp * 2 + hh;
    float qv[HD], kv[HD];
    float qs = 0.f, ks = 0.f;
#pragma unroll
    for (int d = 0; d < HD; ++d) {
      qv[d] = q[(rb + n) * MD + h * HD + d]; qs += qv[d] * qv[d];
      kv[d] = k[(rb + n) * MD + h * HD + d]; ks += kv[d] * kv[d];
    }
    const float qi = rsqrtf(qs + 1e-12f), ki = rsqrtf(ks + 1e-12f);
#pragma unroll
    for (int d = 0; d < HD; ++d) {
      qn[hh][d] = qv[d] * qi;
      kn_lds[wave][h][n][d] = kv[d] * ki;
    }
  }
  __syncthreads();

#pragma unroll
  for (int hh = 0; hh < 2; ++hh) {
    const int h = hp * 2 + hh;
    const float scale = expf(fminf(logit_scale[h], LOG100));
    float sc[KNN], mx = -3.0e38f;
#pragma unroll
    for (int m = 0; m < KNN; ++m) {
      float d0 = 0.f;
#pragma unroll
      for (int d = 0; d < HD; ++d) d0 += qn[hh][d] * kn_lds[wave][h][m][d];
      sc[m] = d0 * scale;
      mx = fmaxf(mx, sc[m]);
    }
    float se = 0.f, sx = 0.f;
#pragma unroll
    for (int m = 0; m < KNN; ++m) {
      const float e = expf(sc[m] - mx);
      se += e; sx += e * x_lds[wave][m];
    }
    const float sval = sx / se;
#pragma unroll
    for (int d = 0; d < HD; ++d)
      o[(rb + n) * MD + h * HD + d] = (__bf16)(sval * Wv[h * HD + d]);
  }
}

// ---------------------------------------------------------------------------
// mnh second layer (per-row scalar) + residual: h3 = h2 + lrelu(t1.W2 + b2)
// ---------------------------------------------------------------------------
__global__ void mnh2_kernel(const __bf16* __restrict__ t1, const float* __restrict__ W2,
                            const float* __restrict__ b2, const float* __restrict__ h2,
                            float* __restrict__ h3) {
  const int r = blockIdx.x * blockDim.x + threadIdx.x;
  const size_t tb = (size_t)r * FF;
  float acc = 0.f;
#pragma unroll 8
  for (int f = 0; f < FF; ++f) acc += (float)t1[tb + f] * W2[f];
  acc += b2[0];
  acc = (acc >= 0.f) ? acc : 0.2f * acc;
  const size_t ob = (size_t)r * MD;
#pragma unroll 8
  for (int i = 0; i < MD; ++i) h3[ob + i] = h2[ob + i] + acc;
}

// ---------------------------------------------------------------------------
// LN3 over NH*MD = 1024 per token; one wave per token, shfl_xor reduction.
// ---------------------------------------------------------------------------
__global__ __launch_bounds__(256) void ln3_kernel(const float* __restrict__ h3,
                                                  const float* __restrict__ g3,
                                                  const float* __restrict__ b3,
                                                  __bf16* __restrict__ y0) {
  const int wave = threadIdx.x >> 5;
  const int lane = threadIdx.x & 31;
  const int tok  = blockIdx.x * 8 + wave;
  const float* row = h3 + (size_t)tok * 1024;
  float s = 0.f, ss = 0.f;
#pragma unroll
  for (int i = 0; i < 32; ++i) {
    const float v = row[lane + i * 32];
    s += v; ss += v * v;
  }
#pragma unroll
  for (int off = 16; off > 0; off >>= 1) {
    s  += __shfl_xor(s, off, 32);
    ss += __shfl_xor(ss, off, 32);
  }
  const float m = s * (1.0f / 1024.0f);
  const float var = ss * (1.0f / 1024.0f) - m * m;
  const float rstd = rsqrtf(var + 1e-5f);
  const size_t ob = (size_t)tok * 1024;
#pragma unroll
  for (int i = 0; i < 32; ++i) {
    const int c = lane + i * 32;
    y0[ob + c] = (__bf16)((row[c] - m) * rstd * g3[c] + b3[c]);
  }
}

// ---------------------------------------------------------------------------
// Launch
// ---------------------------------------------------------------------------
extern "C" void kernel_launch(void* const* d_in, const int* in_sizes, int n_in,
                              void* d_out, int out_size, void* d_ws, size_t ws_size,
                              hipStream_t stream) {
  (void)in_sizes; (void)n_in; (void)out_size; (void)ws_size;
  const float* x      = (const float*)d_in[0];
  const float* c1     = (const float*)d_in[1];
  const float* c2     = (const float*)d_in[2];
  const float* pe_W1  = (const float*)d_in[3];
  const float* pe_b1  = (const float*)d_in[4];
  const float* pe_W2  = (const float*)d_in[5];
  const float* pe_b2  = (const float*)d_in[6];
  const float* Wq     = (const float*)d_in[7];
  const float* Wk     = (const float*)d_in[8];
  const float* Wv     = (const float*)d_in[9];
  const float* lgt    = (const float*)d_in[10];
  const float* Wo     = (const float*)d_in[11];
  const float* bo     = (const float*)d_in[12];
  const float* g1     = (const float*)d_in[13];
  const float* beta1  = (const float*)d_in[14];
  const float* g2     = (const float*)d_in[15];
  const float* beta2  = (const float*)d_in[16];
  const float* g3     = (const float*)d_in[17];
  const float* beta3  = (const float*)d_in[18];
  const float* mnh_W1 = (const float*)d_in[19];
  const float* mnh_b1 = (const float*)d_in[20];
  const float* mnh_W2 = (const float*)d_in[21];
  const float* mnh_b2 = (const float*)d_in[22];
  const float* mu_W1  = (const float*)d_in[23];
  const float* mu_b1  = (const float*)d_in[24];
  const float* mu_W2  = (const float*)d_in[25];
  const float* mu_b2  = (const float*)d_in[26];

  // ---- workspace layout (lifetime-based region reuse, ~157 MB peak) ----
  char* base = (char*)d_ws;
  size_t off = 0;
  auto take = [&](size_t bytes) {
    size_t o = off;
    off += (bytes + 255) & ~(size_t)255;
    return o;
  };
  constexpr size_t SZ_F32_RMD = (size_t)R_ * MD * 4;   // 33.55 MB
  constexpr size_t SZ_BF_RMD  = (size_t)R_ * MD * 2;   // 16.78 MB
  __bf16* peW2b  = (__bf16*)(base + take((size_t)FF * MD * 2));
  __bf16* Wqb    = (__bf16*)(base + take((size_t)MD * MD * 2));
  __bf16* Wkb    = (__bf16*)(base + take((size_t)MD * MD * 2));
  __bf16* Wob    = (__bf16*)(base + take((size_t)MD * MD * 2));
  __bf16* mnhW1b = (__bf16*)(base + take((size_t)MD * FF * 2));
  __bf16* muW1b  = (__bf16*)(base + take((size_t)1024 * 1024 * 2));
  __bf16* muW2b  = (__bf16*)(base + take((size_t)1024 * 1024 * 2));
  float4* nbr    = (float4*)(base + take((size_t)R_ * 16));
  char* regA = base + take(SZ_F32_RMD);
  char* regB = base + take(SZ_F32_RMD);
  char* regC = base + take(SZ_F32_RMD);
  char* regD = base + take(SZ_BF_RMD);
  char* regE = base + take(SZ_F32_RMD);
  // region aliases (disjoint lifetimes, in launch order)
  float4* cand   = (float4*)regA;                 // KNN candidates
  __bf16* a1     = (__bf16*)regA;                 // PE hidden (bf16, R x 128)
  float*  hobuf  = (float*)regA;                  // o @ Wo + bo
  __bf16* t1     = (__bf16*)regA;                 // mnh hidden (bf16, R x 128)
  float*  pe_pre = (float*)regB;                  // PE output pre-LN
  float*  qbuf   = (float*)regB;                  // q
  float*  h2     = (float*)regB;                  // post-LN2 (persists)
  float*  hbuf   = (float*)regC;                  // post-LN1 h
  float*  h3     = (float*)regC;                  // h2 + mnh scalar
  __bf16* hb     = (__bf16*)regD;                 // h bf16
  __bf16* obuf   = (__bf16*)regD;                 // attention output
  __bf16* y0     = (__bf16*)regD;                 // post-LN3 (TOK x 1024)
  float*  kbuf   = (float*)regE;                  // k
  __bf16* h2b    = (__bf16*)regE;                 // h2 bf16
  __bf16* z1     = (__bf16*)(regE + SZ_BF_RMD);   // mu hidden (TOK x 1024)

  auto pack = [&](const float* src, __bf16* dst, int K, int N) {
    pack_b_bf16<<<(K * N + 255) / 256, 256, 0, stream>>>(src, dst, K, N);
  };

  // 0) weights -> fragment-packed bf16
  pack(pe_W2, peW2b, FF, MD);
  pack(Wq, Wqb, MD, MD);
  pack(Wk, Wkb, MD, MD);
  pack(Wo, Wob, MD, MD);
  pack(mnh_W1, mnhW1b, MD, FF);
  pack(mu_W1, muW1b, 1024, 1024);
  pack(mu_W2, muW2b, 1024, 1024);

  // 1) KNN: single streaming pass over c1/c2 (0.54 GB = the HBM floor)
  knn_stage1<<<(B_ * T_ * NCH) / 256, 256, 0, stream>>>(c1, c2, cand);
  knn_stage2<<<TOK / 256, 256, 0, stream>>>(cand, x, nbr);

  // 2) PE MLP: layer 1 (VALU, K=2) -> layer 2 (WMMA, 131072 x 128 @ 128 x 64)
  pe1_kernel<<<(R_ * FF) / 256, 256, 0, stream>>>(nbr, pe_W1, pe_b1, a1);
  rowgemm<FF, MD, 0, 0><<<R_ / 128, 256, 0, stream>>>(a1, peW2b, pe_b2, pe_pre);

  // 3) LN1 (x broadcast residual) -> h (f32 + bf16)
  ln1_kernel<<<R_ / 256, 256, 0, stream>>>(pe_pre, nbr, g1, beta1, hbuf, hb);

  // 4) q = h@Wq, k = h@Wk (WMMA)
  rowgemm<MD, MD, 0, 0><<<R_ / 128, 256, 0, stream>>>(hb, Wqb, nullptr, qbuf);
  rowgemm<MD, MD, 0, 0><<<R_ / 128, 256, 0, stream>>>(hb, Wkb, nullptr, kbuf);

  // 5) cosine attention over NH=16 (wave per token)
  attn_kernel<<<TOK / 8, 256, 0, stream>>>(qbuf, kbuf, nbr, Wv, lgt, obuf);

  // 6) o @ Wo + bo (WMMA), LN2 residual
  rowgemm<MD, MD, 0, 0><<<R_ / 128, 256, 0, stream>>>(obuf, Wob, bo, hobuf);
  ln2_kernel<<<R_ / 256, 256, 0, stream>>>(hbuf, hobuf, g2, beta2, h2, h2b);

  // 7) mnh MLP: layer 1 (WMMA, lrelu, bf16) then scalar layer 2 + residual
  rowgemm<MD, FF, 1, 1><<<R_ / 128, 256, 0, stream>>>(h2b, mnhW1b, mnh_b1, t1);
  mnh2_kernel<<<R_ / 256, 256, 0, stream>>>(t1, mnh_W2, mnh_b2, h2, h3);

  // 8) LN3 over 1024 per token
  ln3_kernel<<<TOK / 8, 256, 0, stream>>>(h3, g3, beta3, y0);

  // 9) unfold MLP: two 8192x1024 @ 1024x1024 WMMA GEMMs via TDM-staged tiles
  mu_gemm<1, 1><<<TOK / 16, 256, 0, stream>>>(y0, muW1b, mu_b1, z1);
  mu_gemm<1, 0><<<TOK / 16, 256, 0, stream>>>(z1, muW2b, mu_b2, d_out);
}
